// LSTM_360777253221
// MI455X (gfx1250) — matrix operand order repaired and checked by reading, not compiled, not run on hardware
//
#include <hip/hip_runtime.h>
#include <hip/hip_bf16.h>

typedef __attribute__((ext_vector_type(16))) _Float16 v16h;
typedef __attribute__((ext_vector_type(8)))  _Float16 v8h;
typedef __attribute__((ext_vector_type(8)))  float    v8f;

#define B_SZ   256
#define T_SZ   1024
#define IN_SZ  38
#define H_SZ   64
#define G_SZ   256   // 4*H
#define OUT_SZ 38
#define KPAD   64    // padded K for the projection GEMMs

#if __has_builtin(__builtin_amdgcn_tanhf)
__device__ __forceinline__ float fast_tanh(float x) {
    return __builtin_amdgcn_tanhf(x);          // v_tanh_f32 (TRANS)
}
#else
__device__ __forceinline__ float fast_tanh(float x) {
    x = fminf(fmaxf(x, -10.0f), 10.0f);
    float e = __expf(2.0f * x);
    return (e - 1.0f) / (e + 1.0f);
}
#endif
__device__ __forceinline__ float fast_sigmoid(float x) {
    return fmaf(fast_tanh(0.5f * x), 0.5f, 0.5f);
}

__device__ __forceinline__ v8f wmma_f16(v16h a, v16h b, v8f c) {
    return __builtin_amdgcn_wmma_f32_16x16x32_f16(
        /*neg_a=*/false, a, /*neg_b=*/false, b,
        /*c_mod=*/(short)0, c, /*reuse_a=*/false, /*reuse_b=*/false);
}

// ---------------------------------------------------------------------------
// prep_x: f32 (B,T,38) -> f16 [b][t][64] zero-padded
// ---------------------------------------------------------------------------
__global__ __launch_bounds__(256)
void prep_x_kernel(const float* __restrict__ x, _Float16* __restrict__ xh)
{
    long idx = (long)blockIdx.x * 256 + threadIdx.x;   // over B*T*64
    if (idx >= (long)B_SZ * T_SZ * KPAD) return;
    int  d = (int)(idx & (KPAD - 1));
    long r = idx >> 6;
    xh[idx] = (d < IN_SZ) ? (_Float16)x[r * IN_SZ + d] : (_Float16)0.0f;
}

// ---------------------------------------------------------------------------
// prep_w: W (256 x Kin) f32 -> Wp f16 [256][64] zero-padded; bias = b_ih+b_hh
// ---------------------------------------------------------------------------
__global__ __launch_bounds__(256)
void prep_w_kernel(const float* __restrict__ W, const float* __restrict__ b_ih,
                   const float* __restrict__ b_hh, _Float16* __restrict__ Wp,
                   float* __restrict__ bias, int Kin)
{
    int idx = blockIdx.x * 256 + threadIdx.x;          // over 256*64
    if (idx >= G_SZ * KPAD) return;
    int n = idx >> 6, k = idx & (KPAD - 1);
    Wp[idx] = (k < Kin) ? (_Float16)W[n * Kin + k] : (_Float16)0.0f;
    if (k == 0) bias[n] = b_ih[n] + b_hh[n];
}

// ---------------------------------------------------------------------------
// inproj: xg = in @ Wp^T + bias, all operands pre-padded f16, K = 64.
// Output in WMMA C-fragment order:
//   xg[(((t*16 + btile)*4 + q)*4 + j)*32 + lane][v]  (8 f32/lane, 32B aligned)
// Gate tile id = q + 4*j. Rows t-major: r = t*256 + b.
// in addressed as in[b*st_b + t*st_t + k]  (strides in halves).
// ---------------------------------------------------------------------------
__global__ __launch_bounds__(256)
void inproj_kernel(const _Float16* __restrict__ in, const _Float16* __restrict__ Wp,
                   const float* __restrict__ bias, float* __restrict__ xg,
                   long st_b, long st_t)
{
    const int lane  = threadIdx.x & 31;
    const int wave  = threadIdx.x >> 5;
    const int gid   = blockIdx.x * 8 + wave;
    const int mtile = gid >> 2;            // 0 .. T*B/16 - 1
    const int q     = gid & 3;
    const int r0    = mtile * 16;
    const int t     = r0 >> 8;             // r = t*256 + b
    const int btile = (r0 & 255) >> 4;
    const int row   = lane & 15;
    const int hi    = lane >> 4;
    const int b     = (r0 & 255) + row;

    const _Float16* inrow = in + (long)b * st_b + (long)t * st_t;

    v8f acc[4] = {};
    #pragma unroll
    for (int ks = 0; ks < 2; ++ks) {
        // A fragment (16x32 f16): element e<8 -> K=koff+e, e>=8 -> K=koff+16+(e-8)
        const v8h* ap = (const v8h*)(inrow + ks * 32 + hi * 8);   // 16B aligned
        v8h alo = ap[0];
        v8h ahi = ap[2];
        v16h a;
        #pragma unroll
        for (int e = 0; e < 8; ++e) { a[e] = alo[e]; a[8 + e] = ahi[e]; }

        // B fragments (32x16 f16): lane col = lane&15, e<8 -> K = ks*32 + 8*hi + e, e>=8 -> K = ks*32 + 16 + 8*hi + (e-8)
        #pragma unroll
        for (int j = 0; j < 4; ++j) {
            const int n = (q + 4 * j) * 16 + row;
            const v8h* wp = (const v8h*)(Wp + (long)n * KPAD + ks * 32 + hi * 8);
            v8h blo = wp[0];
            v8h bhi = wp[2];
            v16h bf;
            #pragma unroll
            for (int e = 0; e < 8; ++e) { bf[e] = blo[e]; bf[8 + e] = bhi[e]; }
            acc[j] = wmma_f16(a, bf, acc[j]);
        }
    }
    #pragma unroll
    for (int j = 0; j < 4; ++j) {
        const int col = (q + 4 * j) * 16 + row;
        v8f outv = acc[j] + bias[col];
        float* dst = xg + ((((long)t * 16 + btile) * 4 + q) * 4 + j) * 256 + lane * 8;
        *(v8f*)dst = outv;                 // 32B aligned -> 2x global_store_b128
    }
}

// ---------------------------------------------------------------------------
// scan: one persistent workgroup per 16 batch rows, 4 waves.
// Wave w owns h cols [16w,16w+16) and gate tiles {w, w+4, w+8, w+12}.
// Double-buffered h in LDS (f16, stride 80) -> ONE barrier per timestep.
// c + W_hh fragments live in registers.
// hall != null -> store f16 h for every t (layer 0); hlast != null -> f32 at T-1.
// ---------------------------------------------------------------------------
__global__ __launch_bounds__(128)
void scan_kernel(const float* __restrict__ xg, const float* __restrict__ Whh,
                 _Float16* __restrict__ hall, float* __restrict__ hlast)
{
    const int lane  = threadIdx.x & 31;
    const int w     = threadIdx.x >> 5;   // 0..3, == q of inproj layout
    const int btile = blockIdx.x;
    const int b0    = btile * 16;
    const int row   = lane & 15;
    const int hi    = lane >> 4;

    __shared__ _Float16 hbuf[2][16 * 80]; // double-buffered [m][h col], stride 80

    // W_hh fragments: B[k][n] = Whh[n][k]; lane col n, K = ks*32 + 8*hi + e (+8 for e>=8)
    v16h Bf[4][2];
    #pragma unroll
    for (int j = 0; j < 4; ++j) {
        const int n = (w + 4 * j) * 16 + row;
        const float* wr = Whh + (long)n * H_SZ;
        #pragma unroll
        for (int ks = 0; ks < 2; ++ks)
            #pragma unroll
            for (int e = 0; e < 16; ++e)
                Bf[j][ks][e] = (_Float16)wr[ks * 32 + hi * 8 + e + (e >= 8 ? 8 : 0)];
    }

    for (int i = threadIdx.x; i < 16 * 80; i += 128) hbuf[0][i] = (_Float16)0.0f;
    v8f c = {};
    __syncthreads();

    for (int t = 0; t < T_SZ; ++t) {
        // A fragments from read buffer: 4 aligned 16B runs (ds_load_b128)
        const v8h* hp = (const v8h*)(hbuf[t & 1] + row * 80 + hi * 8);
        v8h r0v = hp[0], r1v = hp[2], r2v = hp[4], r3v = hp[6];
        v16h a0, a1;
        #pragma unroll
        for (int e = 0; e < 8; ++e) {
            a0[e] = r0v[e];  a0[8 + e] = r1v[e];
            a1[e] = r2v[e];  a1[8 + e] = r3v[e];
        }

        const long base_t = ((long)t * 16 + btile) * 4096 + (long)w * 1024;
        v8f acc[4];
        #pragma unroll
        for (int j = 0; j < 4; ++j)
            acc[j] = *(const v8f*)(xg + base_t + j * 256 + lane * 8);

        if (t + 1 < T_SZ) {
            const long base_n = ((long)(t + 1) * 16 + btile) * 4096 + (long)w * 1024;
            #pragma unroll
            for (int j = 0; j < 4; ++j)
                __builtin_prefetch(xg + base_n + j * 256 + lane * 8, 0, 1);
        }

        #pragma unroll
        for (int j = 0; j < 4; ++j) {
            acc[j] = wmma_f16(a0, Bf[j][0], acc[j]);
            acc[j] = wmma_f16(a1, Bf[j][1], acc[j]);
        }

        // LSTM cell: element v -> batch row m = v + 8*hi, h col = 16w + row
        v8f hv;
        #pragma unroll
        for (int v = 0; v < 8; ++v) {
            float ig = fast_sigmoid(acc[0][v]);
            float fg = fast_sigmoid(acc[1][v]);
            float gg = fast_tanh(acc[2][v]);
            float og = fast_sigmoid(acc[3][v]);
            float cn = fg * c[v] + ig * gg;
            c[v] = cn;
            hv[v] = og * fast_tanh(cn);
        }

        const int col = 16 * w + row;
        _Float16* wb = hbuf[(t + 1) & 1];
        #pragma unroll
        for (int v = 0; v < 8; ++v)
            wb[(v + 8 * hi) * 80 + col] = (_Float16)hv[v];

        if (hall) {
            #pragma unroll
            for (int v = 0; v < 8; ++v) {
                const int m = v + 8 * hi;
                hall[(long)t * B_SZ * H_SZ + (long)(b0 + m) * H_SZ + col] = (_Float16)hv[v];
            }
        }
        if (hlast && t == T_SZ - 1) {
            #pragma unroll
            for (int v = 0; v < 8; ++v) {
                const int m = v + 8 * hi;
                hlast[(long)(b0 + m) * H_SZ + col] = hv[v];
            }
        }
        __syncthreads();   // writes to buf[(t+1)&1] visible before next reads
    }
}

// ---------------------------------------------------------------------------
// head: out = relu(h2_last @ W_out^T + b_out)   (256x64)x(64x38) — tiny
// ---------------------------------------------------------------------------
__global__ __launch_bounds__(256)
void head_kernel(const float* __restrict__ h, const float* __restrict__ Wout,
                 const float* __restrict__ bout, float* __restrict__ out)
{
    int idx = blockIdx.x * blockDim.x + threadIdx.x;
    if (idx >= B_SZ * OUT_SZ) return;
    int b = idx / OUT_SZ, o = idx % OUT_SZ;
    float s = bout[o];
    const float* hr = h + (long)b * H_SZ;
    const float* wr = Wout + (long)o * H_SZ;
    #pragma unroll
    for (int k = 0; k < H_SZ; ++k) s += hr[k] * wr[k];
    out[idx] = fmaxf(s, 0.0f);
}

extern "C" void kernel_launch(void* const* d_in, const int* in_sizes, int n_in,
                              void* d_out, int out_size, void* d_ws, size_t ws_size,
                              hipStream_t stream)
{
    const float* x     = (const float*)d_in[0];
    const float* W_ih0 = (const float*)d_in[1];
    const float* W_hh0 = (const float*)d_in[2];
    const float* b_ih0 = (const float*)d_in[3];
    const float* b_hh0 = (const float*)d_in[4];
    const float* W_ih1 = (const float*)d_in[5];
    const float* W_hh1 = (const float*)d_in[6];
    const float* b_ih1 = (const float*)d_in[7];
    const float* b_hh1 = (const float*)d_in[8];
    const float* W_out = (const float*)d_in[9];
    const float* b_out = (const float*)d_in[10];
    float* out = (float*)d_out;

    // Workspace layout (f32 first, then f16; all 16B aligned)
    float*    xg   = (float*)d_ws;                               // 67,108,864 f32 (256 MiB)
    float*    h2l  = xg + (size_t)T_SZ * B_SZ * G_SZ;            // 16,384 f32
    float*    bias = h2l + (size_t)B_SZ * H_SZ;                  // 256 f32
    _Float16* xh   = (_Float16*)(bias + G_SZ);                   // B*T*64 f16 (32 MiB)
    _Float16* h1h  = xh + (size_t)B_SZ * T_SZ * KPAD;            // T*B*64 f16 (32 MiB), [t][b][h]
    _Float16* Wp   = h1h + (size_t)T_SZ * B_SZ * H_SZ;           // 256*64 f16

    const int gA = (T_SZ * B_SZ / 16) * 4 / 8;                   // 8192 blocks
    const int gX = (int)(((long)B_SZ * T_SZ * KPAD + 255) / 256);
    const int gW = (G_SZ * KPAD + 255) / 256;

    // Layer 0
    prep_x_kernel<<<gX, 256, 0, stream>>>(x, xh);
    prep_w_kernel<<<gW, 256, 0, stream>>>(W_ih0, b_ih0, b_hh0, Wp, bias, IN_SZ);
    inproj_kernel<<<gA, 256, 0, stream>>>(xh, Wp, bias, xg,
                                          (long)T_SZ * KPAD, (long)KPAD);
    scan_kernel<<<B_SZ / 16, 128, 0, stream>>>(xg, W_hh0, h1h, nullptr);

    // Layer 1 (h1h is [t][b][h]: st_b = H, st_t = B*H)
    prep_w_kernel<<<gW, 256, 0, stream>>>(W_ih1, b_ih1, b_hh1, Wp, bias, H_SZ);
    inproj_kernel<<<gA, 256, 0, stream>>>(h1h, Wp, bias, xg,
                                          (long)H_SZ, (long)B_SZ * H_SZ);
    scan_kernel<<<B_SZ / 16, 128, 0, stream>>>(xg, W_hh1, nullptr, h2l);

    // Output head
    head_kernel<<<(B_SZ * OUT_SZ + 255) / 256, 256, 0, stream>>>(h2l, W_out, b_out, out);
}
